// SharedINR_17832704213314
// MI455X (gfx1250) — matrix operand order repaired and checked by reading, not compiled, not run on hardware
//
#include <hip/hip_runtime.h>

typedef __attribute__((ext_vector_type(16))) _Float16 v16h;
typedef __attribute__((ext_vector_type(8)))  _Float16 v8h;
typedef __attribute__((ext_vector_type(8)))  float    v8f;

#define OMEGA 30.0f
#define N_PIX 65536
#define D 256

// ---------------------------------------------------------------- helpers
__global__ void f32_to_f16_kernel(const float* __restrict__ src,
                                  _Float16* __restrict__ dst, int n) {
  int i = blockIdx.x * 256 + threadIdx.x;
  if (i < n) dst[i] = (_Float16)src[i];
}

// First SIREN layer: K=2, plain VALU. out[n,o] = sin(30*(coords[n]·W0[o] + b0[o]))
__global__ void siren_first_kernel(const float* __restrict__ coords,
                                   const float* __restrict__ W0,
                                   const float* __restrict__ b0,
                                   _Float16* __restrict__ out) {
  const int n = blockIdx.x;
  const int o = threadIdx.x;
  const float cx = coords[2 * n], cy = coords[2 * n + 1];
  const float v = __sinf(OMEGA * (cx * W0[2 * o] + cy * W0[2 * o + 1] + b0[o]));
  out[n * D + o] = (_Float16)v;
}

// ---------------------------------------------------------------- core GEMM
// Y[n,o] = sin(OMEGA * (sum_d X[n,d] * W[o,d] + bias[o]))
// X: [N,256] f16 row-major, W: [256,256] f16 row-major (o = row), Y: [N,256] f16
// Block: 256 threads = 8 waves (wave32), 128 rows of X per block.
// - A strip 16x256 per wave resident in 64 VGPRs
// - W staged to LDS in two 128-row halves (64 KB) with CDNA5 async
//   memory->LDS copies (global_load_async_to_lds_b128 / ASYNCcnt)
// - two independent WMMA accumulator chains to pipeline the XDL
__global__ __launch_bounds__(256) void sine_gemm_kernel(
    const _Float16* __restrict__ X, const _Float16* __restrict__ W,
    const float* __restrict__ bias, _Float16* __restrict__ Y) {
  __shared__ alignas(32) _Float16 sW[128 * D];  // 64 KB

  const int tid  = threadIdx.x;
  const int wave = tid >> 5;          // 0..7
  const int lane = tid & 31;
  const int rowBase = blockIdx.x * 128 + wave * 16;
  const int m  = lane & 15;           // A-matrix row within strip
  const int hi = lane >> 4;           // 0: lanes 0-15, 1: lanes 16-31

  // ISA aperture rule: low 32 bits of a generic LDS pointer == LDS byte address
  const unsigned ldsW = (unsigned)(uintptr_t)(void*)sW;

  // --- load A strip (16 rows x 256 K) into registers, contiguous K order (same as B):
  // lane group 0 holds K kf*32+0..15, group 1 holds kf*32+16..31
  v16h a[8];
  {
    const _Float16* xrow = X + (size_t)(rowBase + m) * D;
#pragma unroll
    for (int kf = 0; kf < 8; ++kf) {
      v8h lo = *(const v8h*)(xrow + kf * 32 + hi * 16);
      v8h hh = *(const v8h*)(xrow + kf * 32 + hi * 16 + 8);
      a[kf] = __builtin_shufflevector(lo, hh, 0, 1, 2, 3, 4, 5, 6, 7,
                                      8, 9, 10, 11, 12, 13, 14, 15);
    }
  }

  const int rb = rowBase + hi * 8;    // C-layout: VGPR r holds row rb + r

  for (int half = 0; half < 2; ++half) {
    __syncthreads();                  // all readers done before LDS overwrite
    {
      // async-stage W rows [half*128, half*128+128): 64 KB.
      // 256 threads x 16 transfers x 16 B, direct memory->LDS (no VGPR bounce).
      const unsigned long long gbase =
          (unsigned long long)(uintptr_t)(W + (size_t)half * 128 * D);
#pragma unroll
      for (int i = 0; i < 16; ++i) {
        const unsigned byteOff = (unsigned)(tid + i * 256) * 16u;
        const unsigned lds_addr = ldsW + byteOff;
        const unsigned long long gaddr = gbase + byteOff;
        asm volatile("global_load_async_to_lds_b128 %0, %1, off"
                     :: "v"(lds_addr), "v"(gaddr)
                     : "memory");
      }
      asm volatile("s_wait_asynccnt 0x0" ::: "memory");
    }
    __syncthreads();

    for (int ntl = 0; ntl < 8; ++ntl) {
      const int oloc = ntl * 16 + (lane & 15);  // B column = output neuron
      const int o    = half * 128 + oloc;
      // two independent accumulator chains (even/odd K fragments)
      v8f acc0 = {};
      v8f acc1 = {};
#pragma unroll
      for (int kp = 0; kp < 4; ++kp) {
        // B layout: lane = column, 16 consecutive K per lane
        // (lanes 0-15: K kf*32+0..15, lanes 16-31: K kf*32+16..31)
        v16h b0 = *(const v16h*)(sW + oloc * D + (2 * kp + 0) * 32 + hi * 16);
        acc0 = __builtin_amdgcn_wmma_f32_16x16x32_f16(
            false, a[2 * kp + 0], false, b0, (short)0, acc0, false, false);
        v16h b1 = *(const v16h*)(sW + oloc * D + (2 * kp + 1) * 32 + hi * 16);
        acc1 = __builtin_amdgcn_wmma_f32_16x16x32_f16(
            false, a[2 * kp + 1], false, b1, (short)0, acc1, false, false);
      }
      const float bv = bias[o];
#pragma unroll
      for (int r = 0; r < 8; ++r) {
        const float v = __sinf(OMEGA * ((acc0[r] + acc1[r]) + bv));
        Y[(size_t)(rb + r) * D + o] = (_Float16)v;
      }
    }
  }
}

// ---------------------------------------------------------------- 256 -> 3 head
// out[n,c] = sum_d H[n,d]*Wf[c,d] + bf[c]; one wave per row, shfl_xor reduce.
__global__ __launch_bounds__(256) void final_linear_kernel(
    const _Float16* __restrict__ H, const float* __restrict__ Wf,
    const float* __restrict__ bf, float* __restrict__ out) {
  const int wave = threadIdx.x >> 5;
  const int lane = threadIdx.x & 31;
  const int n = blockIdx.x * 8 + wave;
  float s0 = 0.f, s1 = 0.f, s2 = 0.f;
#pragma unroll
  for (int i = 0; i < 8; ++i) {
    const int d = lane + 32 * i;
    const float h = (float)H[(size_t)n * D + d];
    s0 += h * Wf[0 * D + d];
    s1 += h * Wf[1 * D + d];
    s2 += h * Wf[2 * D + d];
  }
#pragma unroll
  for (int off = 16; off > 0; off >>= 1) {
    s0 += __shfl_xor(s0, off, 32);
    s1 += __shfl_xor(s1, off, 32);
    s2 += __shfl_xor(s2, off, 32);
  }
  if (lane == 0) {
    out[(size_t)n * 3 + 0] = s0 + bf[0];
    out[(size_t)n * 3 + 1] = s1 + bf[1];
    out[(size_t)n * 3 + 2] = s2 + bf[2];
  }
}

// ---------------------------------------------------------------- launch
extern "C" void kernel_launch(void* const* d_in, const int* in_sizes, int n_in,
                              void* d_out, int out_size, void* d_ws, size_t ws_size,
                              hipStream_t stream) {
  const float* coords = (const float*)d_in[0];
  const float* enc_W0 = (const float*)d_in[1];
  const float* enc_b0 = (const float*)d_in[2];
  const float* enc_W  = (const float*)d_in[3];   // [4,256,256]
  const float* enc_b  = (const float*)d_in[4];   // [4,256]
  const float* dec_W0 = (const float*)d_in[5];   // [10,256,256]
  const float* dec_b0 = (const float*)d_in[6];   // [10,256]
  const float* dec_Wm = (const float*)d_in[7];   // [2,10,256,256]
  const float* dec_bm = (const float*)d_in[8];   // [2,10,256]
  const float* dec_Wf = (const float*)d_in[9];   // [10,3,256]
  const float* dec_bf = (const float*)d_in[10];  // [10,3]
  float* out = (float*)d_out;

  // workspace layout:
  //   [0]                 : 34 f16 weight matrices (enc_W:0..3, dec_W0:4..13,
  //                          dec_Wm layer0:14..23, layer1:24..33)  ~4.5 MB
  //   bufA / bufB / bufC  : f16 activation buffers, 33.5 MB each
  char* ws = (char*)d_ws;
  const size_t MATS = (size_t)D * D;  // 65536
  _Float16* wf16 = (_Float16*)ws;
  const size_t wbytes = 34 * MATS * sizeof(_Float16);
  _Float16* bufA = (_Float16*)(ws + wbytes);
  _Float16* bufB = bufA + (size_t)N_PIX * D;
  _Float16* bufC = bufB + (size_t)N_PIX * D;

  // 1) weights -> f16
  {
    const int n1 = 4 * (int)MATS;
    f32_to_f16_kernel<<<(n1 + 255) / 256, 256, 0, stream>>>(enc_W, wf16, n1);
    const int n2 = 10 * (int)MATS;
    f32_to_f16_kernel<<<(n2 + 255) / 256, 256, 0, stream>>>(dec_W0, wf16 + 4 * MATS, n2);
    const int n3 = 20 * (int)MATS;
    f32_to_f16_kernel<<<(n3 + 255) / 256, 256, 0, stream>>>(dec_Wm, wf16 + 14 * MATS, n3);
  }

  // 2) first sine layer (K=2)
  siren_first_kernel<<<N_PIX, 256, 0, stream>>>(coords, enc_W0, enc_b0, bufA);

  const int GB = N_PIX / 128;  // 512 blocks

  // 3) encoder hidden sine layers (4x) : bufA <-> bufB ping-pong
  _Float16* src = bufA;
  _Float16* dst = bufB;
  for (int i = 0; i < 4; ++i) {
    sine_gemm_kernel<<<GB, 256, 0, stream>>>(src, wf16 + (size_t)i * MATS,
                                             enc_b + i * D, dst);
    _Float16* t = src; src = dst; dst = t;
  }
  _Float16* feat = src;  // == bufA after 4 swaps (features, kept live)

  // 4) 10 decoders, sequential, reusing bufB/bufC
  for (int k = 0; k < 10; ++k) {
    sine_gemm_kernel<<<GB, 256, 0, stream>>>(feat, wf16 + (size_t)(4 + k) * MATS,
                                             dec_b0 + k * D, bufB);
    sine_gemm_kernel<<<GB, 256, 0, stream>>>(bufB, wf16 + (size_t)(14 + k) * MATS,
                                             dec_bm + (0 * 10 + k) * D, bufC);
    sine_gemm_kernel<<<GB, 256, 0, stream>>>(bufC, wf16 + (size_t)(24 + k) * MATS,
                                             dec_bm + (1 * 10 + k) * D, bufB);
    final_linear_kernel<<<N_PIX / 8, 256, 0, stream>>>(
        bufB, dec_Wf + (size_t)k * 3 * D, dec_bf + k * 3,
        out + (size_t)k * N_PIX * 3);
  }
}
